// fea_66331474920148
// MI455X (gfx1250) — matrix operand order repaired and checked by reading, not compiled, not run on hardware
//
#include <hip/hip_runtime.h>
#include <math.h>

// ---------------------------------------------------------------------------
// FourierCrossAttention forward for MI455X (gfx1250, wave32, WMMA).
// B=4, L=4096, H=8, E=64, MODES=64, HE=H*E=512.
// Pipeline:
//   K1 init_basis      : bf16 DFT bases (fwd: 128x4096 = [cos;-sin] rows,
//                        inv: 4096x128 = alpha*scale*[cos|-sin] rows)
//   K2 transpose_cvt   : q,k (B,L,HE) f32 -> qT,kT (B,HE,L) bf16
//   K3 dft_gemm (x2)   : per (b,h) 64x4096 @ 4096x128 bf16 WMMA -> Qft/Kft f32
//   K4 freq_core       : per (b,h) complex G=Q^T K, ctanh, V=K G^T, weights -> Y bf16
//   K5 inv_gemm        : 2048x128 @ 128x4096 bf16 WMMA -> out f32 (B,H,O,L)
// Each GEMM wave computes a 16x64 strip with 4 independent accumulators:
// 4 independent v_wmma per k-step (hides XDL latency), 1 A-frag : 4 B-frag loads.
// ---------------------------------------------------------------------------

typedef __attribute__((ext_vector_type(16))) __bf16 v16bf;
typedef __attribute__((ext_vector_type(8)))  __bf16 v8bf;
typedef __attribute__((ext_vector_type(8)))  float  v8f;

#define NB   4
#define NL   4096
#define NH   8
#define NE   64
#define NM   64
#define NHE  512
#define NBH  32   // B*H

union AFrag { v16bf v; v8bf h[2]; };

#define WMMA_BF16(A, Bf, C) \
    __builtin_amdgcn_wmma_f32_16x16x32_bf16(false, (A), false, (Bf), (short)0, (C), false, false)

// ---------------------------------------------------------------------------
// K1: basis init. idx -> (t, m). Exact phase reduction via (m*t) mod 4096.
// fwd[m][t]      = cos(th),  fwd[64+m][t]   = -sin(th)          (rfft)
// inv[t][m]      = a*cos(th), inv[t][64+m]  = -a*sin(th)
//   a = alpha_m / (L * IN_CH * OUT_CH) = 2^-30 (m==0) or 2^-29 (bf16-exact)
// ---------------------------------------------------------------------------
__global__ void init_basis(__bf16* __restrict__ fwd, __bf16* __restrict__ inv) {
    int idx = blockIdx.x * 256 + threadIdx.x;      // 0 .. 4096*64-1
    int m = idx & 63;
    int t = idx >> 6;
    float ph = 6.28318530717958647692f * (float)((m * t) & 4095) * (1.0f / 4096.0f);
    float s, c;
    sincosf(ph, &s, &c);
    fwd[(size_t)m        * NL + t] = (__bf16)c;
    fwd[(size_t)(64 + m) * NL + t] = (__bf16)(-s);
    float a = (m == 0) ? 9.31322574615478515625e-10f   // 2^-30
                       : 1.86264514923095703125e-9f;   // 2^-29
    inv[(size_t)t * 128 + m]      = (__bf16)(a * c);
    inv[(size_t)t * 128 + 64 + m] = (__bf16)(-a * s);
}

// ---------------------------------------------------------------------------
// K2: (B,L,HE) f32 -> (B,HE,L) bf16 via 64x64 LDS tiles. grid.z: 0..3 q, 4..7 k.
// ---------------------------------------------------------------------------
__global__ void transpose_cvt(const float* __restrict__ q, const float* __restrict__ k,
                              __bf16* __restrict__ qT, __bf16* __restrict__ kT) {
    __shared__ float tile[64][65];
    int z = blockIdx.z;
    const float* src = (z < 4) ? q : k;
    __bf16*      dst = (z < 4) ? qT : kT;
    int b  = z & 3;
    int t0 = blockIdx.x * 64;
    int c0 = blockIdx.y * 64;
    int tx = threadIdx.x;   // 0..63
    int ty = threadIdx.y;   // 0..3
#pragma unroll
    for (int j = 0; j < 16; ++j) {
        int row = ty + 4 * j;
        tile[row][tx] = src[((size_t)b * NL + t0 + row) * NHE + c0 + tx];
    }
    __syncthreads();
#pragma unroll
    for (int j = 0; j < 16; ++j) {
        int row = ty + 4 * j;
        dst[((size_t)b * NHE + c0 + row) * NL + t0 + tx] = (__bf16)tile[tx][row];
    }
}

// ---------------------------------------------------------------------------
// K3: per (b,h) GEMM: Xft[e][n] = sum_t XT[e][t] * fwdB[n][t],  M=64,N=128,K=4096.
// Fragment loads use the same contiguous k order for A and B (k = 16*half + s):
//   A 16x32: lanes 0-15 K 0..15, lanes 16-31 K 16..31 (row = lane&15)
//   B 32x16: lanes 0-15 K 0..15, lanes 16-31 K 16..31 (col = lane&15), B^T row-major
// grid (2, 32); block 128 = 4 waves (wave = one 16-row M tile x 64-col strip).
// ---------------------------------------------------------------------------
__global__ void dft_gemm(const __bf16* __restrict__ XT, const __bf16* __restrict__ fwdB,
                         float* __restrict__ Xft) {
    int bh   = blockIdx.y;
    int lane = threadIdx.x & 31;
    int wv   = threadIdx.x >> 5;
    int half = lane >> 4;

    const __bf16* Arow = XT + ((size_t)bh * 64 + wv * 16 + (lane & 15)) * NL;
    int ahalf = half * 16;
    int colBase = blockIdx.x * 64;
    const __bf16* Brow = fwdB + (size_t)(colBase + (lane & 15)) * NL;
    int bhalf = half * 16;

    v8f acc0 = {}, acc1 = {}, acc2 = {}, acc3 = {};
    for (int kk = 0; kk < NL; kk += 32) {
        __builtin_prefetch(Arow + kk + 256, 0, 1);
        AFrag a;
        a.h[0] = *(const v8bf*)(Arow + kk + ahalf);
        a.h[1] = *(const v8bf*)(Arow + kk + ahalf + 8);
        v16bf b0 = *(const v16bf*)(Brow + 0 * 16 * NL + kk + bhalf);
        v16bf b1 = *(const v16bf*)(Brow + 1 * 16 * NL + kk + bhalf);
        v16bf b2 = *(const v16bf*)(Brow + 2 * 16 * NL + kk + bhalf);
        v16bf b3 = *(const v16bf*)(Brow + 3 * 16 * NL + kk + bhalf);
        acc0 = WMMA_BF16(a.v, b0, acc0);
        acc1 = WMMA_BF16(a.v, b1, acc1);
        acc2 = WMMA_BF16(a.v, b2, acc2);
        acc3 = WMMA_BF16(a.v, b3, acc3);
    }
    // C 16x16 f32: VGPR r -> row r (lanes<16) / 8+r (lanes>=16); col = lane&15
    float* C = Xft + (size_t)bh * 64 * 128;
    int crow = wv * 16 + half * 8;
    int ccol = colBase + (lane & 15);
#pragma unroll
    for (int r = 0; r < 8; ++r) {
        C[(size_t)(crow + r) * 128 + ccol +  0] = acc0[r];
        C[(size_t)(crow + r) * 128 + ccol + 16] = acc1[r];
        C[(size_t)(crow + r) * 128 + ccol + 32] = acc2[r];
        C[(size_t)(crow + r) * 128 + ccol + 48] = acc3[r];
    }
}

// ---------------------------------------------------------------------------
// K4: frequency-domain core, one 256-thread block per (b,h).
// 64KB LDS, overlaid: Q,K -> (G over Q) -> (V over K). Register-staged so the
// overlay is safe across __syncthreads. G stored transposed (Gt[y][x]) so all
// hot LDS reads are broadcast or stride-1 (conflict-free).
// ---------------------------------------------------------------------------
__global__ void freq_core(const float* __restrict__ Qft, const float* __restrict__ Kft,
                          const float* __restrict__ W, __bf16* __restrict__ Y) {
    __shared__ float smem[16384];           // 64 KB
    float* Qre = smem;                      // 4096 each
    float* Qim = smem + 4096;
    float* Kre = smem + 8192;
    float* Kim = smem + 12288;
    float* Gtre = smem;                     // overlays Q after step 2
    float* Gtim = smem + 4096;
    float* Vre  = smem + 8192;              // overlays K after step 4
    float* Vim  = smem + 12288;

    int bh  = blockIdx.x;
    int h   = bh & 7;
    int tid = threadIdx.x;

    const float* Qp = Qft + (size_t)bh * 8192;
    const float* Kp = Kft + (size_t)bh * 8192;
    for (int i = tid; i < 4096; i += 256) {
        int e = i >> 6, m = i & 63;
        Qre[i] = Qp[e * 128 + m];
        Qim[i] = Qp[e * 128 + 64 + m];
        Kre[i] = Kp[e * 128 + m];
        Kim[i] = Kp[e * 128 + 64 + m];
    }
    __syncthreads();

    // step 2: G[x][y] = sum_e Q[e][x] * K[e][y]   (complex, no conjugate)
    float ar[16], ai[16];
    {
        int y  = tid & 63;
        int x0 = tid >> 6;
#pragma unroll 4
        for (int j = 0; j < 16; ++j) {
            int x = x0 + 4 * j;
            float sr = 0.f, si = 0.f;
            for (int e = 0; e < 64; ++e) {
                float qr = Qre[e * 64 + x], qi = Qim[e * 64 + x];   // broadcast
                float kr = Kre[e * 64 + y], ki = Kim[e * 64 + y];   // stride-1
                sr += qr * kr - qi * ki;
                si += qr * ki + qi * kr;
            }
            ar[j] = sr; ai[j] = si;
        }
    }
    __syncthreads();                         // Q dead; write G (transposed) over Q
    {
        int y  = tid & 63;
        int x0 = tid >> 6;
#pragma unroll
        for (int j = 0; j < 16; ++j) {
            int x = x0 + 4 * j;
            Gtre[y * 64 + x] = ar[j];
            Gtim[y * 64 + x] = ai[j];
        }
    }
    __syncthreads();

    // step 3: complex tanh in place: tanh(a+ib) = (t(1+u^2) + i u(1-t^2)) / (1+t^2 u^2)
    for (int i = tid; i < 4096; i += 256) {
        float a = Gtre[i], b = Gtim[i];
        float t = tanhf(a), u = tanf(b);
        float t2 = t * t, u2 = u * u;
        float inv = 1.f / (1.f + t2 * u2);
        Gtre[i] = t * (1.f + u2) * inv;
        Gtim[i] = u * (1.f - t2) * inv;
    }
    __syncthreads();

    // step 4: V[e][x] = sum_y G[x][y] * K[e][y]
    {
        int x  = tid & 63;
        int e0 = tid >> 6;
#pragma unroll 4
        for (int j = 0; j < 16; ++j) {
            int e = e0 + 4 * j;
            float sr = 0.f, si = 0.f;
            for (int y = 0; y < 64; ++y) {
                float gr = Gtre[y * 64 + x], gi = Gtim[y * 64 + x]; // stride-1
                float kr = Kre[e * 64 + y], ki = Kim[e * 64 + y];   // broadcast
                sr += gr * kr - gi * ki;
                si += gr * ki + gi * kr;
            }
            ar[j] = sr; ai[j] = si;
        }
    }
    __syncthreads();                         // K dead; write V over K
    {
        int x  = tid & 63;
        int e0 = tid >> 6;
#pragma unroll
        for (int j = 0; j < 16; ++j) {
            int e = e0 + 4 * j;
            Vre[e * 64 + x] = ar[j];
            Vim[e * 64 + x] = ai[j];
        }
    }
    __syncthreads();

    // step 5: Y[o][m] = sum_e V[e][m] * w[h][e][o][m]  (complex64 weights)
    {
        int m  = tid & 63;
        int o0 = tid >> 6;
#pragma unroll 4
        for (int j = 0; j < 16; ++j) {
            int o = o0 + 4 * j;
            const float* wp = W + 2 * ((((size_t)h * 64) * 64 + o) * 64 + m);
            float sr = 0.f, si = 0.f;
            for (int e = 0; e < 64; ++e) {
                float wr = wp[0], wi = wp[1];
                wp += 2 * 64 * 64;                     // e stride (floats)
                float vr = Vre[e * 64 + m], vi = Vim[e * 64 + m];
                sr += vr * wr - vi * wi;
                si += vr * wi + vi * wr;
            }
            size_t row = (size_t)bh * 64 + o;
            Y[row * 128 + m]      = (__bf16)sr;
            Y[row * 128 + 64 + m] = (__bf16)si;
        }
    }
}

// ---------------------------------------------------------------------------
// K5: inverse GEMM. out[row][t] = sum_k Y[row][k] * invB[t][k], M=2048,N=4096,K=128.
// grid (16, 128); block 128 = 4 waves; each wave: 16 rows x 64 cols, 4 accs.
// ---------------------------------------------------------------------------
__global__ void inv_gemm(const __bf16* __restrict__ Y, const __bf16* __restrict__ invB,
                         float* __restrict__ out) {
    int lane = threadIdx.x & 31;
    int wv   = threadIdx.x >> 5;
    int half = lane >> 4;
    int rowBase = blockIdx.y * 16;
    int colBase = blockIdx.x * 256 + wv * 64;

    const __bf16* Arow = Y    + (size_t)(rowBase + (lane & 15)) * 128;
    const __bf16* Brow = invB + (size_t)(colBase + (lane & 15)) * 128;
    int ahalf = half * 16;
    int bhalf = half * 16;

    v8f acc0 = {}, acc1 = {}, acc2 = {}, acc3 = {};
#pragma unroll
    for (int kk = 0; kk < 128; kk += 32) {
        AFrag a;
        a.h[0] = *(const v8bf*)(Arow + kk + ahalf);
        a.h[1] = *(const v8bf*)(Arow + kk + ahalf + 8);
        v16bf b0 = *(const v16bf*)(Brow + 0 * 16 * 128 + kk + bhalf);
        v16bf b1 = *(const v16bf*)(Brow + 1 * 16 * 128 + kk + bhalf);
        v16bf b2 = *(const v16bf*)(Brow + 2 * 16 * 128 + kk + bhalf);
        v16bf b3 = *(const v16bf*)(Brow + 3 * 16 * 128 + kk + bhalf);
        acc0 = WMMA_BF16(a.v, b0, acc0);
        acc1 = WMMA_BF16(a.v, b1, acc1);
        acc2 = WMMA_BF16(a.v, b2, acc2);
        acc3 = WMMA_BF16(a.v, b3, acc3);
    }
    int crow = rowBase + half * 8;
    int ccol = colBase + (lane & 15);
#pragma unroll
    for (int r = 0; r < 8; ++r) {
        out[(size_t)(crow + r) * NL + ccol +  0] = acc0[r];
        out[(size_t)(crow + r) * NL + ccol + 16] = acc1[r];
        out[(size_t)(crow + r) * NL + ccol + 32] = acc2[r];
        out[(size_t)(crow + r) * NL + ccol + 48] = acc3[r];
    }
}

// ---------------------------------------------------------------------------
// Workspace layout (bytes):
//   fwd basis  128*4096*2  =  1 MB   @ 0
//   inv basis  4096*128*2  =  1 MB   @ 1 MB
//   qT         4*512*4096*2= 16 MB   @ 2 MB
//   kT                       16 MB   @ 18 MB
//   Qft        32*64*128*4 =  1 MB   @ 34 MB
//   Kft                       1 MB   @ 35 MB
//   Ybig       2048*128*2  = 0.5 MB  @ 36 MB      total ~36.5 MB
// ---------------------------------------------------------------------------
extern "C" void kernel_launch(void* const* d_in, const int* in_sizes, int n_in,
                              void* d_out, int out_size, void* d_ws, size_t ws_size,
                              hipStream_t stream) {
    (void)in_sizes; (void)n_in; (void)out_size; (void)ws_size;
    const float* q = (const float*)d_in[0];
    const float* k = (const float*)d_in[1];
    // d_in[2] = v : unused by the reference forward
    const float* w = (const float*)d_in[3];          // complex64 interleaved
    float* out = (float*)d_out;

    char* ws = (char*)d_ws;
    const size_t MB = 1024u * 1024u;
    __bf16* fwdB = (__bf16*)(ws + 0);
    __bf16* invB = (__bf16*)(ws + 1 * MB);
    __bf16* qT   = (__bf16*)(ws + 2 * MB);
    __bf16* kT   = (__bf16*)(ws + 18 * MB);
    float*  Qft  = (float*) (ws + 34 * MB);
    float*  Kft  = (float*) (ws + 35 * MB);
    __bf16* Ybig = (__bf16*)(ws + 36 * MB);

    init_basis<<<1024, 256, 0, stream>>>(fwdB, invB);
    transpose_cvt<<<dim3(NL / 64, NHE / 64, 2 * NB), dim3(64, 4), 0, stream>>>(q, k, qT, kT);
    dft_gemm<<<dim3(2, NBH), 128, 0, stream>>>(qT, fwdB, Qft);
    dft_gemm<<<dim3(2, NBH), 128, 0, stream>>>(kT, fwdB, Kft);
    freq_core<<<NBH, 256, 0, stream>>>(Qft, Kft, w, Ybig);
    inv_gemm<<<dim3(NL / 256, 2048 / 16), 128, 0, stream>>>(Ybig, invB, out);
}
